// Seq2Seq_35811437314824
// MI455X (gfx1250) — compile-verified, matrix operand order checked by reading, not run on hardware
//
#include <hip/hip_runtime.h>
#include <hip/hip_bf16.h>

// ---------------------------------------------------------------------------
// Seq2Seq LSTM (B=64, T=128, E=256, H=512, V=10000) for gfx1250 / MI455X.
// All matmuls via v_wmma_f32_16x16x32_f16 (wave32 WMMA), f16 data / f32 acc.
// LDS tiles filled with GLOBAL_LOAD_ASYNC_TO_LDS_B128 (ASYNCcnt path) when
// the toolchain exposes the builtin, else classic load+ds_store staging.
// ---------------------------------------------------------------------------

#define BB   64      // batch
#define TT   128     // sequence length
#define EE   256     // embed dim
#define HH   512     // hidden
#define VV   10000   // vocab
#define GG   (4*HH)  // gate width 2048
#define SROWS ((TT-1)*BB)   // 8128 decoder output rows

typedef __attribute__((ext_vector_type(16))) _Float16 v16h;
typedef __attribute__((ext_vector_type(8)))  _Float16 v8h;
typedef __attribute__((ext_vector_type(8)))  float    v8f;

union FragH { v16h v; _Float16 h[16]; };

// --- async copy-to-LDS (gfx1250 ASYNCcnt path), guarded so compile is safe --
#if __has_builtin(__builtin_amdgcn_global_load_async_to_lds_b128) && \
    __has_builtin(__builtin_amdgcn_s_wait_asynccnt)
#define USE_ASYNC_LDS 1
#else
#define USE_ASYNC_LDS 0
#endif

#if USE_ASYNC_LDS
// Builtin signature (from compiler diagnostic): payload type is a GCC-style
// 16-byte int vector; arg0 = AS(1) pointer (printed as "__device__ *"),
// arg1 = AS(3) pointer, then imm offset, imm cpol.
typedef int async_b128_t __attribute__((vector_size(16)));
typedef __attribute__((address_space(1))) async_b128_t* async_gptr_t;
typedef __attribute__((address_space(3))) async_b128_t* async_lptr_t;
#endif

__device__ __forceinline__ void lds_fill16(void* lds, const void* g)
{
#if USE_ASYNC_LDS
    // Flat LDS addresses carry the LDS byte offset in the low 32 bits
    // (ISA 10.2 aperture mapping), so integer casts produce valid AS(1)/AS(3)
    // pointers without pointer-address-space conversions.
    __builtin_amdgcn_global_load_async_to_lds_b128(
        (async_gptr_t)(unsigned long long)(size_t)g,
        (async_lptr_t)(unsigned)(size_t)lds,
        /*offset=*/0, /*cpol=*/0);
#else
    *(v8h*)lds = *(const v8h*)g;
#endif
}
__device__ __forceinline__ void lds_fill_wait()
{
#if USE_ASYNC_LDS
    __builtin_amdgcn_s_wait_asynccnt(0);
#endif
}

__device__ __forceinline__ unsigned hash_u32(unsigned s) {
    s ^= s >> 17; s *= 0xed5ad4bbu;
    s ^= s >> 11; s *= 0xac4c1b51u;
    s ^= s >> 15; s *= 0x31848babu;
    s ^= s >> 14;
    return s;
}
__device__ __forceinline__ float gumbel_hash(unsigned s) {
    unsigned r = hash_u32(s);
    float u = (float)(r >> 8) * (1.0f / 16777216.0f) + 1e-7f;
    return -__logf(-__logf(u));
}
__device__ __forceinline__ float sigmoidf(float x) {
    return 1.0f / (1.0f + __expf(-x));
}

// ---------------------------------------------------------------------------
// Pack f32 weights -> f16, K-block swizzled layout Wp[(k/32)*N*32 + n*32 + k%32]
// Combined [Wih ; Whh] along K so gates = [x,h] @ Wcomb. For Wout pass Kh=0.
// ---------------------------------------------------------------------------
__global__ void pack_weights(const float* __restrict__ Wih,
                             const float* __restrict__ Whh,
                             _Float16* __restrict__ Wp,
                             int Kx, int Kh, int N)
{
    const size_t total = (size_t)(Kx + Kh) * (size_t)N;
    for (size_t i = (size_t)blockIdx.x * blockDim.x + threadIdx.x;
         i < total; i += (size_t)gridDim.x * blockDim.x) {
        int kk = (int)(i & 31);
        size_t r = i >> 5;
        int n  = (int)(r % N);
        int kb = (int)(r / N);
        int k  = kb * 32 + kk;
        float v = (k < Kx) ? Wih[(size_t)n * Kx + k]
                           : Whh[(size_t)n * Kh + (k - Kx)];
        Wp[i] = (_Float16)v;
    }
}

// ---------------------------------------------------------------------------
// Embedding gather + f32 -> f16.  out[t][b][e] = emb[ids[b][t]][e]
// ---------------------------------------------------------------------------
__global__ void embed_gather(const float* __restrict__ emb,
                             const int* __restrict__ ids,
                             _Float16* __restrict__ out, int Edim)
{
    int t = blockIdx.x, b = blockIdx.y;
    int tok = ids[b * TT + t];
    if (tok < 0) tok = 0; if (tok >= VV) tok = VV - 1;
    const float* src = emb + (size_t)tok * Edim;
    _Float16* dst = out + ((size_t)t * BB + b) * Edim;
    for (int e = threadIdx.x; e < Edim; e += blockDim.x)
        dst[e] = (_Float16)src[e];
}

// ---------------------------------------------------------------------------
// One LSTM timestep for the whole batch.
// Grid: 32 blocks (hidden chunks of 16) x 128 threads (4 waves = 4 M-tiles).
// Each wave computes the 4 gate tiles (i,f,g,o) for its 16 batch rows x 16
// hidden units via WMMA over K = Kx + H, then does the cell update.
// z = [x, h_prev] double-buffered through LDS per 32-wide K slab
// (one barrier per slab; a buffer is re-staged only two iterations later,
//  after an intervening barrier, so WAR is safe).
// ---------------------------------------------------------------------------
__global__ void __launch_bounds__(128)
lstm_step(const _Float16* __restrict__ x,     // [B][Kx]
          const _Float16* __restrict__ hin,   // [B][H]
          float*          __restrict__ c,     // [B][H] in/out (chunk-owned)
          _Float16*       __restrict__ hout,  // [B][H]
          const _Float16* __restrict__ Wp,    // [(Kx+H)/32][4H][32] f16
          const float*    __restrict__ bias,  // [4H]
          int Kx)
{
    const int K = Kx + HH;
    __shared__ _Float16 zt[2][64 * 32];       // 2 x 4 KB activation slabs

    const int tid  = threadIdx.x;
    const int wave = tid >> 5;                // M-tile (batch rows 16*wave..)
    const int lane = tid & 31;
    const int half = lane >> 4;
    const int ln   = lane & 15;
    const int m0   = wave * 16;
    const int j0   = blockIdx.x * 16;         // hidden chunk base

    v8f acc[4];
#pragma unroll
    for (int q = 0; q < 4; ++q)
#pragma unroll
        for (int v = 0; v < 8; ++v) acc[q][v] = 0.0f;

    const int nkb = K >> 5;
    for (int ib = 0; ib < nkb; ++ib) {
        const int kb = ib * 32;
        _Float16* buf = zt[ib & 1];

        // --- stage z[0:64][kb:kb+32] into LDS (each thread: 16 halves) ---
        {
            int r  = tid >> 1;
            int ch = (tid & 1) * 16;
            int kk = kb + ch;
            const _Float16* src = (kk < Kx) ? (x + (size_t)r * Kx + kk)
                                            : (hin + (size_t)r * HH + (kk - Kx));
            lds_fill16(&buf[r * 32 + ch],     src);
            lds_fill16(&buf[r * 32 + ch + 8], src + 8);
        }
        lds_fill_wait();
        __syncthreads();

        // --- A fragment (16x32 f16, documented wave32 layout) from LDS ---
        FragH Af;
        *(v8h*)&Af.h[0] = *(const v8h*)&buf[(m0 + ln) * 32 + half * 8];
        *(v8h*)&Af.h[8] = *(const v8h*)&buf[(m0 + ln) * 32 + 16 + half * 8];

        // --- 4 gate B fragments + WMMA ---
#pragma unroll
        for (int q = 0; q < 4; ++q) {
            int n = q * HH + j0 + ln;
            const _Float16* wp = Wp + ((size_t)(kb >> 5) * GG + n) * 32;
            __builtin_prefetch(wp + (size_t)GG * 32, 0, 1); // next K slab
            FragH Bf;
            *(v8h*)&Bf.h[0] = *(const v8h*)(wp + half * 8);
            *(v8h*)&Bf.h[8] = *(const v8h*)(wp + 16 + half * 8);
            acc[q] = __builtin_amdgcn_wmma_f32_16x16x32_f16(
                false, Af.v, false, Bf.v, (short)0, acc[q], false, false);
        }
    }

    // --- epilogue: gates -> (c,h). C/D layout: col = ln, row = half*8 + v ---
    const int j = j0 + ln;
    const float bi = bias[0 * HH + j], bf = bias[1 * HH + j];
    const float bg = bias[2 * HH + j], bo = bias[3 * HH + j];
#pragma unroll
    for (int v = 0; v < 8; ++v) {
        int m = m0 + half * 8 + v;
        float gi = sigmoidf(acc[0][v] + bi);
        float gf = sigmoidf(acc[1][v] + bf);
        float gg = tanhf(acc[2][v] + bg);
        float go = sigmoidf(acc[3][v] + bo);
        float cn = gf * c[(size_t)m * HH + j] + gi * gg;
        c[(size_t)m * HH + j] = cn;
        hout[(size_t)m * HH + j] = (_Float16)(go * tanhf(cn));
    }
}

// ---------------------------------------------------------------------------
// Batched output projection (8128 x 10000 x 512) fused with Gumbel-argmax.
// Grid: 508 blocks x 256 threads (8 waves). Block owns 16 rows; waves stripe
// over the 625 column tiles; 16 WMMA K-iters per tile from an LDS A-tile.
// ---------------------------------------------------------------------------
__global__ void __launch_bounds__(256)
out_proj_argmax(const _Float16* __restrict__ A,    // [S][512] decoder h1
                const _Float16* __restrict__ Wp,   // [512/32][10000][32] f16
                const float*    __restrict__ bout, // [10000]
                int*            __restrict__ amax) // [S]
{
    __shared__ _Float16 at[16 * 512];   // 16 KB A tile
    __shared__ float    rv[256 * 8];
    __shared__ int      ri[256 * 8];

    const int tid = threadIdx.x;
    const int r0  = blockIdx.x * 16;

    {   // stage 16 rows x 512 halves (1024 x v8h)
        const v8h* src = (const v8h*)(A + (size_t)r0 * 512);
        v8h* dst = (v8h*)at;
        for (int i = tid; i < 1024; i += 256) lds_fill16(&dst[i], &src[i]);
        lds_fill_wait();
    }
    __syncthreads();

    const int wave = tid >> 5, lane = tid & 31;
    const int half = lane >> 4, ln = lane & 15;

    float bestv[8]; int besti[8];
#pragma unroll
    for (int v = 0; v < 8; ++v) { bestv[v] = -1e30f; besti[v] = 0; }

    for (int nt = wave; nt < VV / 16; nt += 8) {
        v8f acc;
#pragma unroll
        for (int v = 0; v < 8; ++v) acc[v] = 0.0f;
#pragma unroll 8
        for (int kb = 0; kb < 512; kb += 32) {
            FragH Af, Bf;
            *(v8h*)&Af.h[0] = *(const v8h*)&at[ln * 512 + kb + half * 8];
            *(v8h*)&Af.h[8] = *(const v8h*)&at[ln * 512 + kb + 16 + half * 8];
            const _Float16* wp = Wp + ((size_t)(kb >> 5) * VV + nt * 16 + ln) * 32;
            *(v8h*)&Bf.h[0] = *(const v8h*)(wp + half * 8);
            *(v8h*)&Bf.h[8] = *(const v8h*)(wp + 16 + half * 8);
            acc = __builtin_amdgcn_wmma_f32_16x16x32_f16(
                false, Af.v, false, Bf.v, (short)0, acc, false, false);
        }
        const int col = nt * 16 + ln;
        const float bb = bout[col];
#pragma unroll
        for (int v = 0; v < 8; ++v) {
            int row = r0 + half * 8 + v;
            // temperature scaling is argmax-invariant; gumbel via hash
            float val = acc[v] + bb + gumbel_hash((unsigned)(row * 10007 + col));
            if (val > bestv[v]) { bestv[v] = val; besti[v] = col; }
        }
    }
#pragma unroll
    for (int v = 0; v < 8; ++v) { rv[tid * 8 + v] = bestv[v]; ri[tid * 8 + v] = besti[v]; }
    __syncthreads();

    if (tid < 16) {             // reduce 128 candidates per row
        int m = tid, hm = m >> 3, vm = m & 7;
        float bv = -1e30f; int bi = 0;
        for (int w = 0; w < 8; ++w)
            for (int l = 0; l < 16; ++l) {
                int lidx = w * 32 + hm * 16 + l;
                float xv = rv[lidx * 8 + vm];
                if (xv > bv) { bv = xv; bi = ri[lidx * 8 + vm]; }
            }
        amax[r0 + m] = bi;
    }
}

// ---------------------------------------------------------------------------
// Mask bookkeeping: idx = exclusive cumsum(mask) per batch; map[t'] = k such
// that mask[b,k+1]>0 and idx[b,k+1]==t' (the scatter target inversion).
// ---------------------------------------------------------------------------
__global__ void prep_indices(const int* __restrict__ mask,
                             int* __restrict__ idx, int* __restrict__ map)
{
    int b = blockIdx.x;
    if (threadIdx.x != 0) return;
    int cacc = 0;
    for (int t = 0; t < TT; ++t) { idx[b * TT + t] = cacc; cacc += (mask[b * TT + t] > 0); }
    for (int t = 0; t < TT; ++t) map[b * TT + t] = -1;
    for (int k = 0; k < TT - 1; ++k)
        if (mask[b * TT + k + 1] > 0) {
            int w = idx[b * TT + k + 1];
            if (w < TT) map[b * TT + w] = k;
        }
}

// ---------------------------------------------------------------------------
// Final assembly: every output row is a one-hot (straight-through y_hard).
// row = b*T + t. masked -> argmax row (decoder or hashed random base);
// unmasked -> one-hot(input id).
// ---------------------------------------------------------------------------
__global__ void fill_onehot(const int* __restrict__ ids,
                            const int* __restrict__ mask,
                            const int* __restrict__ idx,
                            const int* __restrict__ map,
                            const int* __restrict__ amax,
                            float* __restrict__ out)
{
    int row = blockIdx.x;
    int b = row / TT, t = row % TT;
    int hot;
    if (mask[b * TT + t] > 0) {
        int tp = idx[b * TT + t];
        if (tp > TT - 1) tp = TT - 1;
        int k = map[b * TT + tp];
        if (k >= 0) hot = amax[k * BB + b];
        else        hot = (int)(hash_u32((unsigned)(tp * BB + b) * 2654435761u + 12345u) % VV);
    } else {
        hot = ids[b * TT + t];
        if (hot < 0) hot = 0; if (hot >= VV) hot = VV - 1;
    }
    float4* o4 = (float4*)(out + (size_t)row * VV);
    const int n4 = VV / 4, h4 = hot >> 2, hsub = hot & 3;
    for (int i = threadIdx.x; i < n4; i += blockDim.x) {
        float4 z = {0.f, 0.f, 0.f, 0.f};
        if (i == h4) ((float*)&z)[hsub] = 1.0f;
        o4[i] = z;
    }
}

// ---------------------------------------------------------------------------
// Host orchestration
// ---------------------------------------------------------------------------
extern "C" void kernel_launch(void* const* d_in, const int* in_sizes, int n_in,
                              void* d_out, int out_size, void* d_ws, size_t ws_size,
                              hipStream_t stream)
{
    const int*   input_ids = (const int*)  d_in[0];
    const int*   attn_mask = (const int*)  d_in[1];
    const float* enc_embed = (const float*)d_in[3];
    const float* enc_Wih0  = (const float*)d_in[4];
    const float* enc_Whh0  = (const float*)d_in[5];
    const float* enc_b0    = (const float*)d_in[6];
    const float* enc_Wih1  = (const float*)d_in[7];
    const float* enc_Whh1  = (const float*)d_in[8];
    const float* enc_b1    = (const float*)d_in[9];
    const float* dec_embed = (const float*)d_in[10];
    const float* dec_Wih0  = (const float*)d_in[11];
    const float* dec_Whh0  = (const float*)d_in[12];
    const float* dec_b0    = (const float*)d_in[13];
    const float* dec_Wih1  = (const float*)d_in[14];
    const float* dec_Whh1  = (const float*)d_in[15];
    const float* dec_b1    = (const float*)d_in[16];
    const float* dec_Wout  = (const float*)d_in[17];
    const float* dec_bout  = (const float*)d_in[18];
    float* out = (float*)d_out;

    // --- workspace carve-up (256B aligned) ---
    size_t off = 0;
    auto carve = [&](size_t bytes) -> void* {
        void* p = (char*)d_ws + off;
        off += (bytes + 255) & ~(size_t)255;
        return p;
    };
    const size_t BH = (size_t)BB * HH;
    _Float16* Wp_e0 = (_Float16*)carve((size_t)(EE + HH) * GG * 2);  // 3.0 MB
    _Float16* Wp_e1 = (_Float16*)carve((size_t)(HH + HH) * GG * 2);  // 4.0 MB
    _Float16* Wp_d0 = (_Float16*)carve((size_t)(EE + HH) * GG * 2);
    _Float16* Wp_d1 = (_Float16*)carve((size_t)(HH + HH) * GG * 2);
    _Float16* Wp_o  = (_Float16*)carve((size_t)HH * VV * 2);         // 10 MB
    _Float16* enc_x = (_Float16*)carve((size_t)TT * BB * EE * 2);
    _Float16* dec_x = (_Float16*)carve((size_t)(TT - 1) * BB * EE * 2);
    _Float16* ys0   = (_Float16*)carve((size_t)TT * BH * 2);
    _Float16* ysd0  = (_Float16*)carve((size_t)(TT - 1) * BH * 2);
    _Float16* ysd1  = (_Float16*)carve((size_t)(TT - 1) * BH * 2);
    _Float16* hZ    = (_Float16*)carve(BH * 2);
    _Float16* hA    = (_Float16*)carve(BH * 2);
    _Float16* hB    = (_Float16*)carve(BH * 2);
    float*    cA    = (float*)   carve(BH * 4);
    float*    cB    = (float*)   carve(BH * 4);
    int*      amax  = (int*)     carve((size_t)SROWS * 4);
    int*      idxb  = (int*)     carve((size_t)BB * TT * 4);
    int*      mapb  = (int*)     carve((size_t)BB * TT * 4);
    (void)ws_size; (void)n_in; (void)in_sizes; (void)out_size;

    // --- weight packing (f32 -> f16, K-swizzled) ---
    pack_weights<<<1536, 256, 0, stream>>>(enc_Wih0, enc_Whh0, Wp_e0, EE, HH, GG);
    pack_weights<<<2048, 256, 0, stream>>>(enc_Wih1, enc_Whh1, Wp_e1, HH, HH, GG);
    pack_weights<<<1536, 256, 0, stream>>>(dec_Wih0, dec_Whh0, Wp_d0, EE, HH, GG);
    pack_weights<<<2048, 256, 0, stream>>>(dec_Wih1, dec_Whh1, Wp_d1, HH, HH, GG);
    pack_weights<<<2048, 256, 0, stream>>>(dec_Wout, dec_Wout, Wp_o, HH, 0, VV);

    // --- embeddings ---
    embed_gather<<<dim3(TT, BB), 256, 0, stream>>>(enc_embed, input_ids, enc_x, EE);
    embed_gather<<<dim3(TT - 1, BB), 256, 0, stream>>>(dec_embed, input_ids, dec_x, EE);

    // --- zero initial states (every call: determinism) ---
    (void)hipMemsetAsync(hZ, 0, BH * 2, stream);
    (void)hipMemsetAsync(cA, 0, BH * 4, stream);
    (void)hipMemsetAsync(cB, 0, BH * 4, stream);

    // --- encoder layer 0 (h history stored in ys0) ---
    for (int t = 0; t < TT; ++t) {
        const _Float16* hin = (t == 0) ? hZ : (ys0 + (size_t)(t - 1) * BH);
        lstm_step<<<32, 128, 0, stream>>>(enc_x + (size_t)t * BB * EE, hin, cA,
                                          ys0 + (size_t)t * BH, Wp_e0, enc_b0, EE);
    }
    // --- encoder layer 1 (ping-pong; final h lands in hB since T-1 is odd) ---
    for (int t = 0; t < TT; ++t) {
        const _Float16* hin = (t == 0) ? hZ : (((t - 1) & 1) ? hB : hA);
        _Float16* hout = (t & 1) ? hB : hA;
        lstm_step<<<32, 128, 0, stream>>>(ys0 + (size_t)t * BH, hin, cB,
                                          hout, Wp_e1, enc_b1, HH);
    }
    // --- decoder cell 0 (init = enc L0 final state; history in ysd0) ---
    for (int k = 0; k < TT - 1; ++k) {
        const _Float16* hin = (k == 0) ? (ys0 + (size_t)(TT - 1) * BH)
                                       : (ysd0 + (size_t)(k - 1) * BH);
        lstm_step<<<32, 128, 0, stream>>>(dec_x + (size_t)k * BB * EE, hin, cA,
                                          ysd0 + (size_t)k * BH, Wp_d0, dec_b0, EE);
    }
    // --- decoder cell 1 (init = enc L1 final state in hB; history in ysd1) ---
    for (int k = 0; k < TT - 1; ++k) {
        const _Float16* hin = (k == 0) ? hB : (ysd1 + (size_t)(k - 1) * BH);
        lstm_step<<<32, 128, 0, stream>>>(ysd0 + (size_t)k * BH, hin, cB,
                                          ysd1 + (size_t)k * BH, Wp_d1, dec_b1, HH);
    }

    // --- batched output projection + gumbel argmax ---
    out_proj_argmax<<<SROWS / 16, 256, 0, stream>>>(ysd1, Wp_o, dec_bout, amax);

    // --- mask index plumbing + one-hot assembly ---
    prep_indices<<<BB, 32, 0, stream>>>(attn_mask, idxb, mapb);
    fill_onehot<<<BB * TT, 256, 0, stream>>>(input_ids, attn_mask, idxb, mapb, amax, out);
}
